// GRUNetwork_4621384810918
// MI455X (gfx1250) — matrix operand order repaired and checked by reading, not compiled, not run on hardware
//
#include <hip/hip_runtime.h>
#include <math.h>

typedef __attribute__((ext_vector_type(16))) __bf16 v16bf;
typedef __attribute__((ext_vector_type(8)))  float  v8f;

#define SEQ     64
#define BATCH   512
#define IN_DIM  512
#define HID     768
#define G3      (3 * HID)   // 2304
#define TSTEPS  25          // classifier reads h2[24]; later steps are dead work

// ---------------------------------------------------------------------------
// f32 -> bf16 conversion (weights, done once per launch)
// ---------------------------------------------------------------------------
__global__ void gru_cvt_f32_bf16(const float* __restrict__ src,
                                 __bf16* __restrict__ dst, int n) {
  int i = blockIdx.x * blockDim.x + threadIdx.x;
  if (i < n) dst[i] = (__bf16)src[i];
}

__global__ void gru_zero_u32(unsigned int* __restrict__ p, int n) {
  int i = blockIdx.x * blockDim.x + threadIdx.x;
  if (i < n) p[i] = 0u;
}

// ---------------------------------------------------------------------------
// out[m,n] = sum_k A[m,k] * W[n,k] + bias[n]
// M = 512 (batch), N = 2304, K templated (512 or 768), lda == K.
// 256 threads = 8 wave32s arranged 2(M) x 4(N); each wave computes a 32x32
// output tile (2x2 sub-tiles -> 4 accumulators), so each K-chunk issues
// 4 v_wmma per wave off 2 A-fragments + 2 B-fragments (register reuse).
// LDS is double-buffered: stage chunk i+1 while chunk i runs on the WMMA pipe.
// Block tile: 64(M) x 128(N); grid (18, 8).
// ---------------------------------------------------------------------------
template <int K, bool A_IS_F32>
__global__ void __launch_bounds__(256) gemm_bias_wmma(
    const float*  __restrict__ Af,   // used when A_IS_F32
    const __bf16* __restrict__ Ab,   // used when !A_IS_F32
    const __bf16* __restrict__ W,    // [N, K] row-major, K-contiguous
    const float*  __restrict__ bias, // [N]
    float*        __restrict__ out)  // [M, N] f32
{
  constexpr int NCH = K / 32;
  __shared__ __align__(16) __bf16 As[2][64 * 40];   // 64 rows x 32 K (+8 pad)
  __shared__ __align__(16) __bf16 Bs[2][128 * 40];  // 128 rows x 32 K (+8 pad)

  const int tid    = threadIdx.x;
  const int lane   = tid & 31;
  const int wid    = tid >> 5;
  const int waveM  = wid & 1;     // 0..1  -> 32 rows each
  const int waveN  = wid >> 1;    // 0..3  -> 32 cols each
  const int blockM = blockIdx.y * 64;
  const int blockN = blockIdx.x * 128;

  // staging coordinates: A 64x32 (8 elems/thread), B 128x32 (16 elems/thread)
  const int aRow = tid >> 2;        // 0..63
  const int aCol = (tid & 3) * 8;   // 0,8,16,24
  const int bRow = tid >> 1;        // 0..127
  const int bCol = (tid & 1) * 16;  // 0,16

  // CDNA5 WMMA fragment coordinates (ISA 7.12.2, 16-bit A 16x32 / B 32x16)
  const int mA  = lane & 15;
  const int kbA = (lane >> 4) * 8;   // lanes 0-15: K 0-7/16-23; 16-31: 8-15/24-31
  const int kbB = (lane >> 4) * 16;  // unused: B reads use kbA, same K order as A
  const int rA  = (waveM * 32 + mA) * 40;            // A LDS row base (sub-tile 0)
  const int rB  = (waveN * 32 + (lane & 15)) * 40;   // B LDS row base (sub-tile 0)

  auto stage = [&](int ci, int buf) {
    const int k0 = ci * 32;
    if (A_IS_F32) {
      const float* src = Af + (size_t)(blockM + aRow) * K + k0 + aCol;
      const float4 v0 = *(const float4*)(src);
      const float4 v1 = *(const float4*)(src + 4);
      __bf16* d = &As[buf][aRow * 40 + aCol];
      d[0] = (__bf16)v0.x; d[1] = (__bf16)v0.y; d[2] = (__bf16)v0.z; d[3] = (__bf16)v0.w;
      d[4] = (__bf16)v1.x; d[5] = (__bf16)v1.y; d[6] = (__bf16)v1.z; d[7] = (__bf16)v1.w;
    } else {
      *(uint4*)&As[buf][aRow * 40 + aCol] =
          *(const uint4*)(Ab + (size_t)(blockM + aRow) * K + k0 + aCol);
    }
    const __bf16* wr = W + (size_t)(blockN + bRow) * K + k0 + bCol;
    *(uint4*)&Bs[buf][bRow * 40 + bCol]     = *(const uint4*)(wr);
    *(uint4*)&Bs[buf][bRow * 40 + bCol + 8] = *(const uint4*)(wr + 8);
    if (k0 + 32 < K)  // CDNA5 global_prefetch_b8 one chunk ahead
      __builtin_prefetch(wr + 32, 0, 0);
  };

  v8f acc00 = {}, acc01 = {}, acc10 = {}, acc11 = {};

  stage(0, 0);

#pragma unroll
  for (int ci = 0; ci < NCH; ++ci) {
    __syncthreads();                           // buf[ci&1] ready / prev reads done
    if (ci + 1 < NCH) stage(ci + 1, (ci + 1) & 1);

    const __bf16* __restrict__ Al = &As[ci & 1][0];
    const __bf16* __restrict__ Bl = &Bs[ci & 1][0];

    v16bf a0, a1, b0, b1;
#pragma unroll
    for (int j = 0; j < 8; ++j) {
      const int ka = kbA + ((j < 4) ? (2 * j) : (2 * j + 8));
      a0[2 * j]     = Al[rA + ka];
      a0[2 * j + 1] = Al[rA + ka + 1];
      a1[2 * j]     = Al[rA + 16 * 40 + ka];
      a1[2 * j + 1] = Al[rA + 16 * 40 + ka + 1];
      const int kb = kbA + ((j < 4) ? (2 * j) : (2 * j + 8));
      b0[2 * j]     = Bl[rB + kb];
      b0[2 * j + 1] = Bl[rB + kb + 1];
      b1[2 * j]     = Bl[rB + 16 * 40 + kb];
      b1[2 * j + 1] = Bl[rB + 16 * 40 + kb + 1];
    }
    acc00 = __builtin_amdgcn_wmma_f32_16x16x32_bf16(false, a0, false, b0, (short)0, acc00, false, false);
    acc01 = __builtin_amdgcn_wmma_f32_16x16x32_bf16(false, a0, false, b1, (short)0, acc01, false, false);
    acc10 = __builtin_amdgcn_wmma_f32_16x16x32_bf16(false, a1, false, b0, (short)0, acc10, false, false);
    acc11 = __builtin_amdgcn_wmma_f32_16x16x32_bf16(false, a1, false, b1, (short)0, acc11, false, false);
  }

  // ---- epilogue: C layout (VGPR r -> M=r / M=r+8; lane&15 -> N) ----
  const int nG0 = blockN + waveN * 32 + (lane & 15);
  const int mB0 = blockM + waveM * 32 + (lane >> 4) * 8;
  const float bv0 = bias[nG0];
  const float bv1 = bias[nG0 + 16];
#pragma unroll
  for (int r = 0; r < 8; ++r) {
    out[(size_t)(mB0 + r)      * G3 + nG0]      = acc00[r] + bv0;
    out[(size_t)(mB0 + r)      * G3 + nG0 + 16] = acc01[r] + bv1;
    out[(size_t)(mB0 + 16 + r) * G3 + nG0]      = acc10[r] + bv0;
    out[(size_t)(mB0 + 16 + r) * G3 + nG0 + 16] = acc11[r] + bv1;
  }
}

// ---------------------------------------------------------------------------
// Fused GRU gates: r = sig(xr+hr); z = sig(xz+hz); n = tanh(xn + r*hn);
// h' = (1-z)*n + z*h.  Writes f32 state + bf16 mirror for the next WMMA step.
// ---------------------------------------------------------------------------
__global__ void gru_update(const float* __restrict__ xg,
                           const float* __restrict__ hg,
                           float* __restrict__ h,
                           __bf16* __restrict__ hb) {
  int i = blockIdx.x * blockDim.x + threadIdx.x;
  if (i >= BATCH * HID) return;
  int b = i / HID, c = i % HID;
  const float* xr = xg + (size_t)b * G3;
  const float* hr = hg + (size_t)b * G3;
  float r  = 1.0f / (1.0f + __expf(-(xr[c]           + hr[c])));
  float z  = 1.0f / (1.0f + __expf(-(xr[HID + c]     + hr[HID + c])));
  float nn = tanhf(xr[2 * HID + c] + r * hr[2 * HID + c]);
  float hp = h[i];
  float hn = (1.0f - z) * nn + z * hp;
  h[i]  = hn;
  hb[i] = (__bf16)hn;
}

// ---------------------------------------------------------------------------
// Classifier: one wave32 per batch row; shuffle reduction + sigmoid.
// ---------------------------------------------------------------------------
__global__ void gru_classifier(const float* __restrict__ h,
                               const float* __restrict__ wcls,
                               const float* __restrict__ bcls,
                               float* __restrict__ out) {
  int b    = blockIdx.x * 8 + (threadIdx.x >> 5);
  int lane = threadIdx.x & 31;
  float s = 0.0f;
  for (int i = lane; i < HID; i += 32) s += h[(size_t)b * HID + i] * wcls[i];
#pragma unroll
  for (int off = 16; off; off >>= 1) s += __shfl_xor(s, off, 32);
  if (lane == 0) out[b] = 1.0f / (1.0f + __expf(-(s + bcls[0])));
}

// ---------------------------------------------------------------------------
extern "C" void kernel_launch(void* const* d_in, const int* in_sizes, int n_in,
                              void* d_out, int out_size, void* d_ws, size_t ws_size,
                              hipStream_t stream) {
  const float* x     = (const float*)d_in[0];
  const float* Wih0  = (const float*)d_in[1];
  const float* Whh0  = (const float*)d_in[2];
  const float* bih0  = (const float*)d_in[3];
  const float* bhh0  = (const float*)d_in[4];
  const float* Wih1  = (const float*)d_in[5];
  const float* Whh1  = (const float*)d_in[6];
  const float* bih1  = (const float*)d_in[7];
  const float* bhh1  = (const float*)d_in[8];
  const float* Wcls  = (const float*)d_in[9];
  const float* bcls  = (const float*)d_in[10];
  float* out = (float*)d_out;

  // ---- workspace layout (bytes, all 256-aligned) ----
  char* ws = (char*)d_ws;
  const size_t nWih0 = (size_t)G3 * IN_DIM;   // 1,179,648
  const size_t nWhh  = (size_t)G3 * HID;      // 1,769,472
  size_t off = 0;
  __bf16* wih0_bf = (__bf16*)(ws + off); off += nWih0 * 2;
  __bf16* whh0_bf = (__bf16*)(ws + off); off += nWhh  * 2;
  __bf16* wih1_bf = (__bf16*)(ws + off); off += nWhh  * 2;
  __bf16* whh1_bf = (__bf16*)(ws + off); off += nWhh  * 2;
  char*   state0  = ws + off;                                // zeroed region start
  float*  h0_f32  = (float*)(ws + off);  off += (size_t)BATCH * HID * 4;
  __bf16* h0_bf   = (__bf16*)(ws + off); off += (size_t)BATCH * HID * 2;
  float*  h1_f32  = (float*)(ws + off);  off += (size_t)BATCH * HID * 4;
  __bf16* h1_bf   = (__bf16*)(ws + off); off += (size_t)BATCH * HID * 2;
  size_t  stateBytes = (size_t)(ws + off - state0);
  float*  xg = (float*)(ws + off); off += (size_t)BATCH * G3 * 4;
  float*  hg = (float*)(ws + off); off += (size_t)BATCH * G3 * 4;   // total ~27 MB

  // ---- one-time (per launch) weight conversion + state zero ----
  const int CB = 256;
  gru_cvt_f32_bf16<<<(int)((nWih0 + CB - 1) / CB), CB, 0, stream>>>(Wih0, wih0_bf, (int)nWih0);
  gru_cvt_f32_bf16<<<(int)((nWhh  + CB - 1) / CB), CB, 0, stream>>>(Whh0, whh0_bf, (int)nWhh);
  gru_cvt_f32_bf16<<<(int)((nWhh  + CB - 1) / CB), CB, 0, stream>>>(Wih1, wih1_bf, (int)nWhh);
  gru_cvt_f32_bf16<<<(int)((nWhh  + CB - 1) / CB), CB, 0, stream>>>(Whh1, whh1_bf, (int)nWhh);
  {
    int nw = (int)(stateBytes / 4);
    gru_zero_u32<<<(nw + CB - 1) / CB, CB, 0, stream>>>((unsigned int*)state0, nw);
  }

  // ---- 25 interleaved recurrent steps ----
  dim3 gGrid(G3 / 128, BATCH / 64);   // (18, 8)
  dim3 gBlk(256);
  int upBlocks = (BATCH * HID + CB - 1) / CB;

  for (int t = 0; t < TSTEPS; ++t) {
    const float* x_t = x + (size_t)t * BATCH * IN_DIM;
    // layer 0: xg = x_t @ Wih0^T + bih0 ; hg = h0 @ Whh0^T + bhh0
    gemm_bias_wmma<IN_DIM, true ><<<gGrid, gBlk, 0, stream>>>(x_t, nullptr, wih0_bf, bih0, xg);
    gemm_bias_wmma<HID,    false><<<gGrid, gBlk, 0, stream>>>(nullptr, h0_bf, whh0_bf, bhh0, hg);
    gru_update<<<upBlocks, CB, 0, stream>>>(xg, hg, h0_f32, h0_bf);   // h0 is now h1_t
    // layer 1: xg = h1_t @ Wih1^T + bih1 ; hg = h1 @ Whh1^T + bhh1
    gemm_bias_wmma<HID, false><<<gGrid, gBlk, 0, stream>>>(nullptr, h0_bf, wih1_bf, bih1, xg);
    gemm_bias_wmma<HID, false><<<gGrid, gBlk, 0, stream>>>(nullptr, h1_bf, whh1_bf, bhh1, hg);
    gru_update<<<upBlocks, CB, 0, stream>>>(xg, hg, h1_f32, h1_bf);
  }

  // ---- classifier on h2[24] ----
  gru_classifier<<<BATCH / 8, 256, 0, stream>>>(h1_f32, Wcls, bcls, out);
}
